// PairformerModule_1898375545297
// MI455X (gfx1250) — compile-verified, matrix operand order checked by reading, not run on hardware
//
#include <hip/hip_runtime.h>
#include <cstdint>
#include <cstddef>

// ============================================================================
// Pairformer (2 blocks) for MI455X / gfx1250.
// Bandwidth-bound (~220 GFLOP vs 23.3 TB/s HBM): all GEMM operands are f16,
// math runs on v_wmma_f32_16x16x32_f16 with f32 accumulation, epilogues
// (bias/sigmoid/silu/alpha/strided scatter/f16 store) are fused.
// GEMM kernels:
//   k_gemm_big : M%128==0, N%64==0, K%32==0. A/B tiles are DMA'd into
//                double-buffered LDS by the Tensor Data Mover
//                (tensor_load_to_lds + s_wait_tensorcnt; D# pad feature
//                reproduces the 40-half padded row stride) with full
//                compute/DMA overlap. Toolchain probe: 6-arg builtin form.
//                Each wave computes 32x32 via 2x2 fragments = 4 WMMA/K-step.
//   k_gemm_gen : edge shapes (N=4/16/24/32, K=24); vector-or-zero 8-half loads.
// ============================================================================

typedef _Float16 h16;
typedef __attribute__((ext_vector_type(16))) _Float16 v16h;
typedef __attribute__((ext_vector_type(8)))  float    v8f;
typedef __attribute__((ext_vector_type(4)))  unsigned int v4u;
typedef __attribute__((ext_vector_type(4)))  int  i32x4;
typedef __attribute__((ext_vector_type(8)))  int  i32x8;

#if defined(__HIP_DEVICE_COMPILE__) && __has_builtin(__builtin_amdgcn_tensor_load_to_lds)
#define USE_TDM 1
#else
#define USE_TDM 0
#endif

#define NT   256     // tokens (N)
#define DZ   128     // pair channel
#define DS2  384     // single channel
#define NHH  16      // apb heads
#define PHH  4       // triangle attn heads
#define PWW  32      // triangle attn head width
#define INF_ 1e9f
#define EPS_ 1e-5f

static constexpr long RR = (long)NT * NT;   // 65536 pair rows

union HV { v16h v; h16 h[16]; v4u q[2]; };
union FV { v8f  v; float f[8]; };
union H8 { h16 h[8]; v4u q; };

struct GemmP {
  const h16* A; int lda; long sAo, sAi;
  const h16* B; int ldb; long sBo, sBi;
  float* Cf; h16* Ch;
  long crs, ccs, sCo, sCi;
  const float* bias;
  int M, N, K;
  int innerB;
  float alpha;
  int act;   // 0 none, 1 sigmoid, 2 silu
};

#define TK  32
#define LDT (TK + 8)   // 80B LDS row stride: 16B-aligned, bank-friendly

__device__ __forceinline__ void store_tile(const GemmP& g, const FV& a, int mbase,
                                           int nc, long cb, bool guard) {
  if (guard && nc >= g.N) return;
  const float bv = g.bias ? g.bias[nc] : 0.f;
#pragma unroll
  for (int v = 0; v < 8; ++v) {
    const int mr = mbase + v;
    if (guard && mr >= g.M) continue;
    float val = a.f[v] * g.alpha + bv;
    if (g.act == 1)      val = 1.f / (1.f + __expf(-val));
    else if (g.act == 2) val = val / (1.f + __expf(-val));
    const long off = cb + (long)mr * g.crs + (long)nc * g.ccs;
    if (g.Cf) g.Cf[off] = val;
    if (g.Ch) g.Ch[off] = (h16)val;
  }
}

#if USE_TDM
// Issue one TDM descriptor: DMA a rows x 32-half 2D tile (f16, row stride
// stride_h halves) from global into LDS at byte offset lds_off, padding each
// 16-DWORD row with 4 DWORDs -> 40-half LDS row stride (= LDT).
// D# bitfields per CDNA5 ISA.
// Toolchain exposes the 6-arg builtin: (u32x4, i32x8, i32x4, i32x4, i32x8, cpol).
__device__ __forceinline__ void tdm_load_2d(const h16* gaddr, unsigned lds_off,
                                            int rows, long stride_h) {
  const unsigned long long ga = (unsigned long long)(size_t)gaddr;
  v4u g0;
  g0[0] = 1u;                                            // count=1, user mode
  g0[1] = lds_off;                                       // lds_addr
  g0[2] = (unsigned)(ga & 0xffffffffull);                // global_addr[31:0]
  g0[3] = (unsigned)((ga >> 32) & 0x1ffffffull)          // global_addr[56:32]
          | (2u << 30);                                  // type = 2 ("image")
  i32x8 g1;
  g1[0] = (1 << 16)            // data_size = 2 bytes
        | (1 << 20)            // pad_enable
        | (3 << 22)            // pad_interval: 16 DWORDs (one 32-half row)
        | (3 << 25);           // pad_amount: 4 DWORDs (8 halves)
  g1[1] = (int)((TK & 0xffff) << 16);                    // tensor_dim0 lo16 (=32)
  g1[2] = (int)((rows & 0xffff) << 16);                  // dim0 hi16=0 | tensor_dim1 lo16
  g1[3] = (rows >> 16) | (TK << 16);                     // tensor_dim1 hi16 | tile_dim0=32
  g1[4] = rows & 0xffff;                                 // tile_dim1=rows, tile_dim2=0
  g1[5] = (int)(stride_h & 0xffffffffl);                 // tensor_dim0_stride lo32
  g1[6] = (int)((stride_h >> 32) & 0xffffl);             // stride hi16 | dim1_stride lo16=0
  g1[7] = 0;
  i32x4 gz4; gz4[0] = 0; gz4[1] = 0; gz4[2] = 0; gz4[3] = 0;   // groups 2/3 unused (2D)
  i32x8 gz8;
#pragma unroll
  for (int i = 0; i < 8; ++i) gz8[i] = 0;                      // trailing group unused
  __builtin_amdgcn_tensor_load_to_lds(g0, g1, gz4, gz4, gz8, 0);
}
#endif

// ---------------------------------------------------------------------------
// Big kernel: 128x64 tile, no bounds checks. 8 waves; wave (wm,wn) computes a
// 32x32 sub-tile: 2 A fragments x 2 B fragments -> 4 WMMA per K-step.
// ---------------------------------------------------------------------------
__global__ __launch_bounds__(256) void k_gemm_big(GemmP g) {
#if USE_TDM
  __shared__ __align__(16) h16 As[2][128][LDT];   // double-buffered for TDM overlap
  __shared__ __align__(16) h16 Bs[2][64][LDT];
#else
  __shared__ __align__(16) h16 As[1][128][LDT];
  __shared__ __align__(16) h16 Bs[1][64][LDT];
#endif

  const int bz = blockIdx.z;
  const int ob = bz / g.innerB;
  const int ib = bz % g.innerB;
  const h16* A = g.A + ob * g.sAo + ib * g.sAi;
  const h16* B = g.B + ob * g.sBo + ib * g.sBi;

  const int m0 = blockIdx.x * 128;
  const int n0 = blockIdx.y * 64;
  const int tid  = threadIdx.x;
  const int lane = tid & 31;
  const int wave = tid >> 5;
  const int wm   = wave & 3;     // 4 wave-rows, 32 M each
  const int wn   = wave >> 2;    // 2 wave-cols, 32 N each
  const int half_id = lane >> 4;
  const int l16     = lane & 15;

  FV acc[2][2];
#pragma unroll
  for (int i = 0; i < 2; ++i)
#pragma unroll
    for (int j = 0; j < 2; ++j)
#pragma unroll
      for (int v = 0; v < 8; ++v) acc[i][j].f[v] = 0.f;

#if USE_TDM
  const h16* Abase = A + (long)m0 * g.lda;
  const h16* Bbase = B + (long)n0 * g.ldb;
  const unsigned offA0 = (unsigned)(size_t)(const void*)&As[0][0][0];
  const unsigned offA1 = (unsigned)(size_t)(const void*)&As[1][0][0];
  const unsigned offB0 = (unsigned)(size_t)(const void*)&Bs[0][0][0];
  const unsigned offB1 = (unsigned)(size_t)(const void*)&Bs[1][0][0];
  if (wave == 0) {                       // prologue: DMA first K-tile
    tdm_load_2d(Abase, offA0, 128, g.lda);
    tdm_load_2d(Bbase, offB0, 64,  g.ldb);
    __builtin_amdgcn_s_wait_tensorcnt(0);
  }
  __syncthreads();
  int cur = 0;
  for (int kt = 0; kt < g.K; kt += TK) {
    if (wave == 0 && kt + TK < g.K) {    // DMA next tile while computing this one
      tdm_load_2d(Abase + kt + TK, cur ? offA0 : offA1, 128, g.lda);
      tdm_load_2d(Bbase + kt + TK, cur ? offB0 : offB1, 64,  g.ldb);
    }
    HV a0, a1, b0, b1;
    a0.q[0] = *(const v4u*)(&As[cur][wm * 32 + l16][8 * half_id]);
    a0.q[1] = *(const v4u*)(&As[cur][wm * 32 + l16][16 + 8 * half_id]);
    a1.q[0] = *(const v4u*)(&As[cur][wm * 32 + 16 + l16][8 * half_id]);
    a1.q[1] = *(const v4u*)(&As[cur][wm * 32 + 16 + l16][16 + 8 * half_id]);
    b0.q[0] = *(const v4u*)(&Bs[cur][wn * 32 + l16][8 * half_id]);
    b0.q[1] = *(const v4u*)(&Bs[cur][wn * 32 + l16][16 + 8 * half_id]);
    b1.q[0] = *(const v4u*)(&Bs[cur][wn * 32 + 16 + l16][8 * half_id]);
    b1.q[1] = *(const v4u*)(&Bs[cur][wn * 32 + 16 + l16][16 + 8 * half_id]);

    acc[0][0].v = __builtin_amdgcn_wmma_f32_16x16x32_f16(false, a0.v, false, b0.v,
                                                         (short)0, acc[0][0].v, false, false);
    acc[0][1].v = __builtin_amdgcn_wmma_f32_16x16x32_f16(false, a0.v, false, b1.v,
                                                         (short)0, acc[0][1].v, false, false);
    acc[1][0].v = __builtin_amdgcn_wmma_f32_16x16x32_f16(false, a1.v, false, b0.v,
                                                         (short)0, acc[1][0].v, false, false);
    acc[1][1].v = __builtin_amdgcn_wmma_f32_16x16x32_f16(false, a1.v, false, b1.v,
                                                         (short)0, acc[1][1].v, false, false);
    if (wave == 0) __builtin_amdgcn_s_wait_tensorcnt(0);   // next tile landed
    __syncthreads();
    cur ^= 1;
  }
#else
  const int lr = tid >> 2;          // 0..63
  const int lc = (tid & 3) * 8;     // 0,8,16,24
  for (int kt = 0; kt < g.K; kt += TK) {
    *(v4u*)(&As[0][lr][lc])      = *(const v4u*)(A + (long)(m0 + lr) * g.lda + kt + lc);
    *(v4u*)(&As[0][lr + 64][lc]) = *(const v4u*)(A + (long)(m0 + lr + 64) * g.lda + kt + lc);
    *(v4u*)(&Bs[0][lr][lc])      = *(const v4u*)(B + (long)(n0 + lr) * g.ldb + kt + lc);
    __builtin_prefetch(A + (long)(m0 + lr) * g.lda + kt + TK + lc, 0, 1);
    __builtin_prefetch(A + (long)(m0 + lr + 64) * g.lda + kt + TK + lc, 0, 1);
    __builtin_prefetch(B + (long)(n0 + lr) * g.ldb + kt + TK + lc, 0, 1);
    __syncthreads();

    HV a0, a1, b0, b1;
    a0.q[0] = *(const v4u*)(&As[0][wm * 32 + l16][8 * half_id]);
    a0.q[1] = *(const v4u*)(&As[0][wm * 32 + l16][16 + 8 * half_id]);
    a1.q[0] = *(const v4u*)(&As[0][wm * 32 + 16 + l16][8 * half_id]);
    a1.q[1] = *(const v4u*)(&As[0][wm * 32 + 16 + l16][16 + 8 * half_id]);
    b0.q[0] = *(const v4u*)(&Bs[0][wn * 32 + l16][8 * half_id]);
    b0.q[1] = *(const v4u*)(&Bs[0][wn * 32 + l16][16 + 8 * half_id]);
    b1.q[0] = *(const v4u*)(&Bs[0][wn * 32 + 16 + l16][8 * half_id]);
    b1.q[1] = *(const v4u*)(&Bs[0][wn * 32 + 16 + l16][16 + 8 * half_id]);

    acc[0][0].v = __builtin_amdgcn_wmma_f32_16x16x32_f16(false, a0.v, false, b0.v,
                                                         (short)0, acc[0][0].v, false, false);
    acc[0][1].v = __builtin_amdgcn_wmma_f32_16x16x32_f16(false, a0.v, false, b1.v,
                                                         (short)0, acc[0][1].v, false, false);
    acc[1][0].v = __builtin_amdgcn_wmma_f32_16x16x32_f16(false, a1.v, false, b0.v,
                                                         (short)0, acc[1][0].v, false, false);
    acc[1][1].v = __builtin_amdgcn_wmma_f32_16x16x32_f16(false, a1.v, false, b1.v,
                                                         (short)0, acc[1][1].v, false, false);
    __syncthreads();
  }
#endif

  const long cb = ob * g.sCo + ib * g.sCi;
#pragma unroll
  for (int ti = 0; ti < 2; ++ti)
#pragma unroll
    for (int tj = 0; tj < 2; ++tj)
      store_tile(g, acc[ti][tj], m0 + wm * 32 + ti * 16 + half_id * 8,
                 n0 + wn * 32 + tj * 16 + l16, cb, false);
}

// ---------------------------------------------------------------------------
// Generic kernel: 64x64 tile; vector-or-zero 8-half loads (K always %8==0 in
// this model), guarded epilogue. Used only for small edge shapes.
// ---------------------------------------------------------------------------
__global__ __launch_bounds__(256) void k_gemm_gen(GemmP g) {
  __shared__ __align__(16) h16 As[64][LDT];
  __shared__ __align__(16) h16 Bs[64][LDT];

  const int bz = blockIdx.z;
  const int ob = bz / g.innerB;
  const int ib = bz % g.innerB;
  const h16* A = g.A + ob * g.sAo + ib * g.sAi;
  const h16* B = g.B + ob * g.sBo + ib * g.sBi;

  const int m0 = blockIdx.x * 64;
  const int n0 = blockIdx.y * 64;
  const int tid  = threadIdx.x;
  const int lane = tid & 31;
  const int wave = tid >> 5;
  const int wm   = wave & 3;
  const int wn   = wave >> 2;
  const int half_id = lane >> 4;
  const int l16     = lane & 15;

  FV acc0, acc1;
#pragma unroll
  for (int v = 0; v < 8; ++v) { acc0.f[v] = 0.f; acc1.f[v] = 0.f; }

  const int lr = tid >> 2;
  const int lc = (tid & 3) * 8;

  for (int kt = 0; kt < g.K; kt += TK) {
    {
      H8 t;
#pragma unroll
      for (int e = 0; e < 8; ++e) t.h[e] = (h16)0.f;
      const int gr = m0 + lr, gk = kt + lc;
      if (gr < g.M && gk + 8 <= g.K) t.q = *(const v4u*)(A + (long)gr * g.lda + gk);
      *(v4u*)(&As[lr][lc]) = t.q;
    }
    {
      H8 t;
#pragma unroll
      for (int e = 0; e < 8; ++e) t.h[e] = (h16)0.f;
      const int gr = n0 + lr, gk = kt + lc;
      if (gr < g.N && gk + 8 <= g.K) t.q = *(const v4u*)(B + (long)gr * g.ldb + gk);
      *(v4u*)(&Bs[lr][lc]) = t.q;
    }
    __syncthreads();

    HV af, bf0, bf1;
    af.q[0]  = *(const v4u*)(&As[wm * 16 + l16][8 * half_id]);
    af.q[1]  = *(const v4u*)(&As[wm * 16 + l16][16 + 8 * half_id]);
    bf0.q[0] = *(const v4u*)(&Bs[wn * 32 + l16][8 * half_id]);
    bf0.q[1] = *(const v4u*)(&Bs[wn * 32 + l16][16 + 8 * half_id]);
    bf1.q[0] = *(const v4u*)(&Bs[wn * 32 + 16 + l16][8 * half_id]);
    bf1.q[1] = *(const v4u*)(&Bs[wn * 32 + 16 + l16][16 + 8 * half_id]);

    acc0.v = __builtin_amdgcn_wmma_f32_16x16x32_f16(false, af.v, false, bf0.v,
                                                    (short)0, acc0.v, false, false);
    acc1.v = __builtin_amdgcn_wmma_f32_16x16x32_f16(false, af.v, false, bf1.v,
                                                    (short)0, acc1.v, false, false);
    __syncthreads();
  }

  const long cb = ob * g.sCo + ib * g.sCi;
  store_tile(g, acc0, m0 + wm * 16 + half_id * 8, n0 + wn * 32 + l16, cb, true);
  store_tile(g, acc1, m0 + wm * 16 + half_id * 8, n0 + wn * 32 + 16 + l16, cb, true);
}

// ---------------------------------------------------------------------------
// LayerNorm over last dim D, f32 in -> f16 out (feeds WMMA GEMMs)
// ---------------------------------------------------------------------------
__global__ __launch_bounds__(128) void k_layernorm(const float* __restrict__ x,
                                                   const float* __restrict__ gamma,
                                                   const float* __restrict__ beta,
                                                   h16* __restrict__ outh, int D) {
  const long row = blockIdx.x;
  const int tid = threadIdx.x;
  const float* xr = x + row * D;
  float s = 0.f, s2 = 0.f;
  for (int i = tid; i < D; i += 128) { float v = xr[i]; s += v; s2 += v * v; }
  __shared__ float r1[128], r2[128];
  r1[tid] = s; r2[tid] = s2;
  __syncthreads();
  for (int st = 64; st > 0; st >>= 1) {
    if (tid < st) { r1[tid] += r1[tid + st]; r2[tid] += r2[tid + st]; }
    __syncthreads();
  }
  const float mean = r1[0] / (float)D;
  const float var  = r2[0] / (float)D - mean * mean;
  const float rstd = rsqrtf(var + EPS_);
  h16* orow = outh + row * D;
  for (int i = tid; i < D; i += 128)
    orow[i] = (h16)((xr[i] - mean) * rstd * gamma[i] + beta[i]);
}

// ---------------------------------------------------------------------------
// Elementwise / pack / softmax kernels
// ---------------------------------------------------------------------------
__global__ void k_copy_f32(const float* __restrict__ s, float* __restrict__ d, long n) {
  long t = (long)blockIdx.x * 256 + threadIdx.x;
  if (t < n) d[t] = s[t];
}
__global__ void k_cvt_f16(const float* __restrict__ s, h16* __restrict__ d, long n) {
  long t = (long)blockIdx.x * 256 + threadIdx.x;
  if (t < n) d[t] = (h16)s[t];
}
__global__ void k_add(float* __restrict__ z, const float* __restrict__ a, long n) {
  long t = (long)blockIdx.x * 256 + threadIdx.x;
  if (t < n) z[t] += a[t];
}
// z[i,j,:] += a[j,i,:]
__global__ void k_add_swap(float* __restrict__ z, const float* __restrict__ a, int n, int d) {
  long t = (long)blockIdx.x * 256 + threadIdx.x;
  long total = (long)n * n * d;
  if (t >= total) return;
  int c = (int)(t % d); long r = t / d;
  int j = (int)(r % n); int i = (int)(r / n);
  z[t] += a[((long)j * n + i) * d + c];
}
// z += a * sigmoid(g)
__global__ void k_add_gate(float* __restrict__ z, const float* __restrict__ a,
                           const float* __restrict__ g, long n) {
  long t = (long)blockIdx.x * 256 + threadIdx.x;
  if (t < n) z[t] += a[t] * (1.f / (1.f + __expf(-g[t])));
}
// o(f16) *= g(f32, already sigmoid)
__global__ void k_mul_f16(h16* __restrict__ o, const float* __restrict__ g, long n) {
  long t = (long)blockIdx.x * 256 + threadIdx.x;
  if (t < n) o[t] = (h16)((float)o[t] * g[t]);
}
// out = silu(h1) * h2  -> f16
__global__ void k_silu_mul_f16(const float* __restrict__ h1, const float* __restrict__ h2,
                               h16* __restrict__ o, long n) {
  long t = (long)blockIdx.x * 256 + threadIdx.x;
  if (t >= n) return;
  float a = h1[t];
  o[t] = (h16)((a / (1.f + __expf(-a))) * h2[t]);
}
// dst[j,i,:] = src[i,j,:]
__global__ void k_transpose_nnd(const float* __restrict__ src, float* __restrict__ dst,
                                int n, int d) {
  long t = (long)blockIdx.x * 256 + threadIdx.x;
  long total = (long)n * n * d;
  if (t >= total) return;
  int c = (int)(t % d); long r = t / d;
  int j = (int)(r % n); int i = (int)(r / n);
  dst[((long)j * n + i) * d + c] = src[t];
}
// tri-mul gating + mask + transpose to [d][i][k] f16 (swap selects in/outgoing)
__global__ void k_pack_tri(const float* __restrict__ proj, const float* __restrict__ gate,
                           const float* __restrict__ pm, h16* __restrict__ at,
                           h16* __restrict__ bt, int n, int swap) {
  long t = (long)blockIdx.x * 256 + threadIdx.x;
  long total = (long)n * n * DZ;
  if (t >= total) return;
  int d = (int)(t % DZ); long r = t / DZ;
  int v = (int)(r % n); int u = (int)(r / n);
  const long src = r * (2 * DZ);
  const float pmv = pm[r];
  const float pa = proj[src + d]      * (1.f / (1.f + __expf(-gate[src + d])))      * pmv;
  const float pb = proj[src + DZ + d] * (1.f / (1.f + __expf(-gate[src + DZ + d]))) * pmv;
  const long oi = swap ? ((long)d * n * n + (long)v * n + u)
                       : ((long)d * n * n + (long)u * n + v);
  at[oi] = (h16)pa;
  bt[oi] = (h16)pb;
}
// triatt V pack: vt[i][h][w][k] = v[(i*N+k)*128 + h*32+w]
__global__ void k_pack_vt_tri(const h16* __restrict__ v, h16* __restrict__ vt) {
  long t = (long)blockIdx.x * 256 + threadIdx.x;
  const long total = (long)NT * PHH * PWW * NT;
  if (t >= total) return;
  int k = (int)(t % NT); long r = t / NT;
  int w = (int)(r % PWW); r /= PWW;
  int h = (int)(r % PHH); int i = (int)(r / PHH);
  vt[t] = v[((long)i * NT + k) * DZ + h * PWW + w];
}
// apb V pack: vt[h][dd][j] = v[j*384 + h*24+dd]
__global__ void k_pack_vt_apb(const h16* __restrict__ v, h16* __restrict__ vt) {
  long t = (long)blockIdx.x * 256 + threadIdx.x;
  const long total = (long)NHH * 24 * NT;
  if (t >= total) return;
  int j = (int)(t % NT); long r = t / NT;
  int dd = (int)(r % 24); int h = (int)(r / 24);
  vt[t] = v[(long)j * DS2 + h * 24 + dd];
}
// softmax over k for rows [ci][h][j]; adds INF*(pm-1) and triangle bias
__global__ __launch_bounds__(256) void k_softmax_tri(const float* __restrict__ sc,
                                                     h16* __restrict__ pr,
                                                     const float* __restrict__ pm,
                                                     const float* __restrict__ tb,
                                                     int i0, int swapPM) {
  const int row = blockIdx.x;
  const int k = threadIdx.x;
  const int j  = row % NT;
  const int ci = row / (NT * PHH);
  const int i  = i0 + ci;
  const long o = (long)row * NT + k;
  float v = sc[o];
  const float pmv = swapPM ? pm[(long)k * NT + i] : pm[(long)i * NT + k];
  v += INF_ * (pmv - 1.f);
  v += tb[((long)j * NT + k) * PHH + (row / NT) % PHH];
  __shared__ float red[256];
  red[k] = v; __syncthreads();
  for (int st = 128; st > 0; st >>= 1) {
    if (k < st) red[k] = fmaxf(red[k], red[k + st]);
    __syncthreads();
  }
  const float mx = red[0];
  __syncthreads();
  const float e = __expf(v - mx);
  red[k] = e; __syncthreads();
  for (int st = 128; st > 0; st >>= 1) {
    if (k < st) red[k] += red[k + st];
    __syncthreads();
  }
  pr[o] = (h16)(e / red[0]);
}
// softmax over j for rows [h][i]; adds pair bias zb[(i,j),h] and mask bias
__global__ __launch_bounds__(256) void k_softmax_apb(const float* __restrict__ sc,
                                                     h16* __restrict__ pr,
                                                     const float* __restrict__ mask,
                                                     const float* __restrict__ zb) {
  const int row = blockIdx.x;
  const int j = threadIdx.x;
  const int i = row % NT;
  const int h = row / NT;
  const long o = (long)row * NT + j;
  float v = sc[o] + zb[((long)i * NT + j) * NHH + h] + (1.f - mask[j]) * (-INF_);
  __shared__ float red[256];
  red[j] = v; __syncthreads();
  for (int st = 128; st > 0; st >>= 1) {
    if (j < st) red[j] = fmaxf(red[j], red[j + st]);
    __syncthreads();
  }
  const float mx = red[0];
  __syncthreads();
  const float e = __expf(v - mx);
  red[j] = e; __syncthreads();
  for (int st = 128; st > 0; st >>= 1) {
    if (j < st) red[j] += red[j + st];
    __syncthreads();
  }
  pr[o] = (h16)(e / red[0]);
}

// ---------------------------------------------------------------------------
// Host-side orchestration
// ---------------------------------------------------------------------------
struct Lin { const float* w; const float* b; h16* w16; int dout, din; };
struct LNp { const float* b; const float* g; };
struct TriMul { Lin g_in, g_out; LNp ln_in, ln_out; Lin p_in, p_out; };
struct TriAtt { Lin bias; LNp ln; Lin wg, wk, wo, wq, wv; };
struct Apb { LNp ln_s, ln_z; Lin wg, wk, wo, wq, wv, wz; };
struct Trans { Lin fc1, fc2, fc3; LNp ln; int d; };
struct Layer { Apb apb; TriAtt tae, tas; TriMul tmi, tmo; Trans ts, tz; };

struct Ctx {
  hipStream_t st;
  h16 *xf16, *xs16, *ta, *tb, *qb, *kb, *vb, *vt, *ob, *pr;
  h16 *q16, *k16, *v16, *vt16, *o16, *prS;
  float *bufA, *bufB, *tbuf, *tbh, *zb, *scS, *gS;
};

static inline dim3 ew(long n) { return dim3((unsigned)((n + 255) / 256), 1, 1); }

static void gemm(Ctx& c,
                 const h16* A, int lda, long sAo, long sAi,
                 const h16* B, int ldb, long sBo, long sBi,
                 float* Cf, h16* Ch, long crs, long ccs, long sCo, long sCi,
                 const float* bias, int M, int N, int K,
                 int batches, int innerB, float alpha, int act) {
  GemmP g;
  g.A = A; g.lda = lda; g.sAo = sAo; g.sAi = sAi;
  g.B = B; g.ldb = ldb; g.sBo = sBo; g.sBi = sBi;
  g.Cf = Cf; g.Ch = Ch; g.crs = crs; g.ccs = ccs; g.sCo = sCo; g.sCi = sCi;
  g.bias = bias; g.M = M; g.N = N; g.K = K;
  g.innerB = (innerB < 1) ? 1 : innerB; g.alpha = alpha; g.act = act;
  if ((M % 128 == 0) && (N % 64 == 0) && (K % 32 == 0)) {
    dim3 gr(M / 128, N / 64, batches);
    k_gemm_big<<<gr, 256, 0, c.st>>>(g);
  } else {
    dim3 gr((M + 63) / 64, (N + 63) / 64, batches);
    k_gemm_gen<<<gr, 256, 0, c.st>>>(g);
  }
}

static void run_trimul(Ctx& c, float* z, const float* pm, const TriMul& p, bool outgoing) {
  k_layernorm<<<(unsigned)RR, 128, 0, c.st>>>(z, p.ln_in.g, p.ln_in.b, c.xf16, DZ);
  gemm(c, c.xf16, DZ, 0, 0, p.p_in.w16, DZ, 0, 0, c.bufA, nullptr, 2 * DZ, 1, 0, 0,
       nullptr, (int)RR, 2 * DZ, DZ, 1, 1, 1.f, 0);
  gemm(c, c.xf16, DZ, 0, 0, p.g_in.w16, DZ, 0, 0, c.bufB, nullptr, 2 * DZ, 1, 0, 0,
       nullptr, (int)RR, 2 * DZ, DZ, 1, 1, 1.f, 0);
  k_pack_tri<<<ew(RR * DZ), 256, 0, c.st>>>(c.bufA, c.bufB, pm, c.ta, c.tb, NT,
                                            outgoing ? 0 : 1);
  // 128 batched GEMMs over d: t[i,j,d], output scattered as [i][j][d]
  gemm(c, c.ta, NT, 0, RR, c.tb, NT, 0, RR, c.tbuf, nullptr,
       (long)NT * DZ, DZ, 0, 1, nullptr, NT, NT, NT, DZ, DZ, 1.f, 0);
  gemm(c, c.xf16, DZ, 0, 0, p.g_out.w16, DZ, 0, 0, c.bufB, nullptr, DZ, 1, 0, 0,
       nullptr, (int)RR, DZ, DZ, 1, 1, 1.f, 0);
  k_layernorm<<<(unsigned)RR, 128, 0, c.st>>>(c.tbuf, p.ln_out.g, p.ln_out.b, c.xf16, DZ);
  gemm(c, c.xf16, DZ, 0, 0, p.p_out.w16, DZ, 0, 0, c.bufA, nullptr, DZ, 1, 0, 0,
       nullptr, (int)RR, DZ, DZ, 1, 1, 1.f, 0);
  k_add_gate<<<ew(RR * DZ), 256, 0, c.st>>>(z, c.bufA, c.bufB, RR * DZ);
}

static void run_triatt(Ctx& c, float* z, const float* pm, const TriAtt& p, bool starting) {
  const float* zin = z;
  if (!starting) {
    k_transpose_nnd<<<ew(RR * DZ), 256, 0, c.st>>>(z, c.tbuf, NT, DZ);
    zin = c.tbuf;
  }
  k_layernorm<<<(unsigned)RR, 128, 0, c.st>>>(zin, p.ln.g, p.ln.b, c.xf16, DZ);
  gemm(c, c.xf16, DZ, 0, 0, p.wq.w16, DZ, 0, 0, nullptr, c.qb, DZ, 1, 0, 0,
       nullptr, (int)RR, DZ, DZ, 1, 1, 1.f, 0);
  gemm(c, c.xf16, DZ, 0, 0, p.wk.w16, DZ, 0, 0, nullptr, c.kb, DZ, 1, 0, 0,
       nullptr, (int)RR, DZ, DZ, 1, 1, 1.f, 0);
  gemm(c, c.xf16, DZ, 0, 0, p.wv.w16, DZ, 0, 0, nullptr, c.vb, DZ, 1, 0, 0,
       nullptr, (int)RR, DZ, DZ, 1, 1, 1.f, 0);
  gemm(c, c.xf16, DZ, 0, 0, p.bias.w16, DZ, 0, 0, c.tbh, nullptr, PHH, 1, 0, 0,
       nullptr, (int)RR, PHH, DZ, 1, 1, 1.f, 0);
  gemm(c, c.xf16, DZ, 0, 0, p.wg.w16, DZ, 0, 0, c.bufB, nullptr, DZ, 1, 0, 0,
       p.wg.b, (int)RR, DZ, DZ, 1, 1, 1.f, 1);          // sigmoid gate, f32
  k_pack_vt_tri<<<ew(RR * DZ), 256, 0, c.st>>>(c.vb, c.vt);
  const int CH = 64;   // i-chunk: scores fit bufA (67 MB), probs fit tbuf
  for (int i0 = 0; i0 < NT; i0 += CH) {
    gemm(c, c.qb + (long)i0 * NT * DZ, DZ, (long)NT * DZ, PWW,
            c.kb + (long)i0 * NT * DZ, DZ, (long)NT * DZ, PWW,
            c.bufA, nullptr, NT, 1, (long)PHH * NT * NT, (long)NT * NT,
            nullptr, NT, NT, PWW, CH * PHH, PHH, 0.1767766953f, 0);
    k_softmax_tri<<<CH * PHH * NT, NT, 0, c.st>>>(c.bufA, c.pr, pm, c.tbh, i0,
                                                  starting ? 0 : 1);
    gemm(c, c.pr, NT, (long)PHH * NT * NT, (long)NT * NT,
            c.vt + (long)i0 * PHH * PWW * NT, NT, (long)PHH * PWW * NT, (long)PWW * NT,
            nullptr, c.ob + (long)i0 * NT * DZ, DZ, 1, (long)NT * DZ, PWW,
            nullptr, NT, PWW, NT, CH * PHH, PHH, 1.f, 0);
  }
  k_mul_f16<<<ew(RR * DZ), 256, 0, c.st>>>(c.ob, c.bufB, RR * DZ);
  gemm(c, c.ob, DZ, 0, 0, p.wo.w16, DZ, 0, 0, c.bufB, nullptr, DZ, 1, 0, 0,
       p.wo.b, (int)RR, DZ, DZ, 1, 1, 1.f, 0);
  if (starting) k_add<<<ew(RR * DZ), 256, 0, c.st>>>(z, c.bufB, RR * DZ);
  else          k_add_swap<<<ew(RR * DZ), 256, 0, c.st>>>(z, c.bufB, NT, DZ);
}

static void run_transition(Ctx& c, float* x, const Trans& p, h16* xbuf, long rows) {
  const int d = p.d, d4 = 4 * p.d;
  k_layernorm<<<(unsigned)rows, 128, 0, c.st>>>(x, p.ln.g, p.ln.b, xbuf, d);
  const long CHr = (d == DZ) ? 16384 : rows;
  for (long r0 = 0; r0 < rows; r0 += CHr) {
    const int m = (int)((rows - r0 < CHr) ? (rows - r0) : CHr);
    gemm(c, xbuf + r0 * d, d, 0, 0, p.fc1.w16, d, 0, 0, c.bufA, nullptr, d4, 1, 0, 0,
         nullptr, m, d4, d, 1, 1, 1.f, 0);
    gemm(c, xbuf + r0 * d, d, 0, 0, p.fc2.w16, d, 0, 0, c.bufB, nullptr, d4, 1, 0, 0,
         nullptr, m, d4, d, 1, 1, 1.f, 0);
    k_silu_mul_f16<<<ew((long)m * d4), 256, 0, c.st>>>(c.bufA, c.bufB, c.ta, (long)m * d4);
    gemm(c, c.ta, d4, 0, 0, p.fc3.w16, d4, 0, 0, c.bufB, nullptr, d, 1, 0, 0,
         nullptr, m, d, d4, 1, 1, 1.f, 0);
    k_add<<<ew((long)m * d), 256, 0, c.st>>>(x + r0 * d, c.bufB, (long)m * d);
  }
}

static void run_apb(Ctx& c, float* s, float* z, const float* mask, const Apb& p) {
  k_layernorm<<<NT, 128, 0, c.st>>>(s, p.ln_s.g, p.ln_s.b, c.xs16, DS2);
  gemm(c, c.xs16, DS2, 0, 0, p.wq.w16, DS2, 0, 0, nullptr, c.q16, DS2, 1, 0, 0,
       p.wq.b, NT, DS2, DS2, 1, 1, 1.f, 0);
  gemm(c, c.xs16, DS2, 0, 0, p.wk.w16, DS2, 0, 0, nullptr, c.k16, DS2, 1, 0, 0,
       nullptr, NT, DS2, DS2, 1, 1, 1.f, 0);
  gemm(c, c.xs16, DS2, 0, 0, p.wv.w16, DS2, 0, 0, nullptr, c.v16, DS2, 1, 0, 0,
       nullptr, NT, DS2, DS2, 1, 1, 1.f, 0);
  k_pack_vt_apb<<<ew((long)NHH * 24 * NT), 256, 0, c.st>>>(c.v16, c.vt16);
  k_layernorm<<<(unsigned)RR, 128, 0, c.st>>>(z, p.ln_z.g, p.ln_z.b, c.xf16, DZ);
  gemm(c, c.xf16, DZ, 0, 0, p.wz.w16, DZ, 0, 0, c.zb, nullptr, NHH, 1, 0, 0,
       nullptr, (int)RR, NHH, DZ, 1, 1, 1.f, 0);
  // per-head scores, K=24 (zero-padded by the generic tile loader)
  gemm(c, c.q16, DS2, 0, 24, c.k16, DS2, 0, 24, c.scS, nullptr, NT, 1, 0,
       (long)NT * NT, nullptr, NT, NT, 24, NHH, NHH, 0.2041241452f, 0);
  k_softmax_apb<<<NHH * NT, NT, 0, c.st>>>(c.scS, c.prS, mask, c.zb);
  gemm(c, c.prS, NT, 0, (long)NT * NT, c.vt16, NT, 0, (long)24 * NT,
       nullptr, c.o16, DS2, 1, 0, 24, nullptr, NT, 24, NT, NHH, NHH, 1.f, 0);
  gemm(c, c.xs16, DS2, 0, 0, p.wg.w16, DS2, 0, 0, c.gS, nullptr, DS2, 1, 0, 0,
       nullptr, NT, DS2, DS2, 1, 1, 1.f, 1);            // sigmoid gate
  k_mul_f16<<<ew((long)NT * DS2), 256, 0, c.st>>>(c.o16, c.gS, (long)NT * DS2);
  gemm(c, c.o16, DS2, 0, 0, p.wo.w16, DS2, 0, 0, c.bufB, nullptr, DS2, 1, 0, 0,
       nullptr, NT, DS2, DS2, 1, 1, 1.f, 0);
  k_add<<<ew((long)NT * DS2), 256, 0, c.st>>>(s, c.bufB, (long)NT * DS2);
}

extern "C" void kernel_launch(void* const* d_in, const int* in_sizes, int n_in,
                              void* d_out, int out_size, void* d_ws, size_t ws_size,
                              hipStream_t stream) {
  (void)in_sizes; (void)n_in; (void)out_size; (void)ws_size;
  Ctx c; c.st = stream;

  // ---- workspace bump allocator (256B aligned) ----
  char* base = (char*)d_ws;
  size_t off = 0;
  auto alloc = [&](size_t bytes) -> void* {
    void* pp = base + off;
    off = (off + bytes + 255) & ~(size_t)255;
    return pp;
  };

  c.xf16 = (h16*)alloc((size_t)RR * DZ * 2);
  c.xs16 = (h16*)alloc((size_t)NT * DS2 * 2);
  c.bufA = (float*)alloc((size_t)RR * 2 * DZ * 4);   // 67 MB (proj / scores / fc1)
  c.bufB = (float*)alloc((size_t)RR * 2 * DZ * 4);   // 67 MB (gate / fc2 / outs)
  c.ta   = (h16*)alloc((size_t)RR * DZ * 2);
  c.tb   = (h16*)alloc((size_t)RR * DZ * 2);
  c.tbuf = (float*)alloc((size_t)RR * DZ * 4);       // t / z^T; aliased as probs
  c.qb   = (h16*)alloc((size_t)RR * DZ * 2);
  c.kb   = (h16*)alloc((size_t)RR * DZ * 2);
  c.ob   = (h16*)alloc((size_t)RR * DZ * 2);
  c.tbh  = (float*)alloc((size_t)RR * PHH * 4);
  c.vb = c.ta;                 // triatt V reuses trimul f16 scratch
  c.vt = c.tb;
  c.pr = (h16*)c.tbuf;         // softmax probs alias (z^T dead after LN)
  c.q16  = (h16*)alloc((size_t)NT * DS2 * 2);
  c.k16  = (h16*)alloc((size_t)NT * DS2 * 2);
  c.v16  = (h16*)alloc((size_t)NT * DS2 * 2);
  c.o16  = (h16*)alloc((size_t)NT * DS2 * 2);
  c.vt16 = (h16*)alloc((size_t)NHH * 24 * NT * 2);
  c.scS  = (float*)alloc((size_t)NHH * NT * NT * 4);
  c.prS  = (h16*)alloc((size_t)NHH * NT * NT * 2);
  c.zb   = (float*)alloc((size_t)RR * NHH * 4);
  c.gS   = (float*)alloc((size_t)NT * DS2 * 4);

  // ---- parse params (jax pytree order: dict keys sorted at every level) ----
  int cur = 4;
  auto nxt = [&]() { return (const float*)d_in[cur++]; };
  auto mklin = [&](int dout, int din, bool hasb) -> Lin {
    Lin L; L.dout = dout; L.din = din;
    L.b = hasb ? nxt() : nullptr;
    L.w = nxt();
    long n = (long)dout * din;
    L.w16 = (h16*)alloc((size_t)n * sizeof(h16));
    k_cvt_f16<<<ew(n), 256, 0, stream>>>(L.w, L.w16, n);
    return L;
  };
  auto mkln = [&]() -> LNp { LNp P; P.b = nxt(); P.g = nxt(); return P; };
  auto mktrimul = [&]() -> TriMul {
    TriMul m;
    m.g_in = mklin(2 * DZ, DZ, false); m.g_out = mklin(DZ, DZ, false);
    m.ln_in = mkln(); m.ln_out = mkln();
    m.p_in = mklin(2 * DZ, DZ, false); m.p_out = mklin(DZ, DZ, false);
    return m;
  };
  auto mktriatt = [&]() -> TriAtt {
    TriAtt a;
    a.bias = mklin(PHH, DZ, false); a.ln = mkln();
    a.wg = mklin(DZ, DZ, true); a.wk = mklin(DZ, DZ, false);
    a.wo = mklin(DZ, DZ, true); a.wq = mklin(DZ, DZ, false);
    a.wv = mklin(DZ, DZ, false);
    return a;
  };
  auto mkapb = [&]() -> Apb {
    Apb a;
    a.ln_s = mkln(); a.ln_z = mkln();
    a.wg = mklin(DS2, DS2, false); a.wk = mklin(DS2, DS2, false);
    a.wo = mklin(DS2, DS2, false); a.wq = mklin(DS2, DS2, true);
    a.wv = mklin(DS2, DS2, false); a.wz = mklin(NHH, DZ, false);
    return a;
  };
  auto mktrans = [&](int d) -> Trans {
    Trans t; t.d = d;
    t.fc1 = mklin(4 * d, d, false); t.fc2 = mklin(4 * d, d, false);
    t.fc3 = mklin(d, 4 * d, false); t.ln = mkln();
    return t;
  };

  Layer Ls[2];
  for (int l = 0; l < 2; ++l) {
    Layer L;
    L.apb = mkapb(); L.tae = mktriatt(); L.tas = mktriatt();
    L.tmi = mktrimul(); L.tmo = mktrimul();
    L.ts = mktrans(DS2); L.tz = mktrans(DZ);
    Ls[l] = L;
  }

  // ---- state lives directly in d_out: [s | z] ----
  float* s_cur = (float*)d_out;
  float* z_cur = s_cur + (long)NT * DS2;
  k_copy_f32<<<ew((long)NT * DS2), 256, 0, stream>>>((const float*)d_in[0], s_cur,
                                                     (long)NT * DS2);
  k_copy_f32<<<ew(RR * DZ), 256, 0, stream>>>((const float*)d_in[1], z_cur, RR * DZ);
  const float* mask = (const float*)d_in[2];
  const float* pm   = (const float*)d_in[3];

  for (int l = 0; l < 2; ++l) {
    run_trimul(c, z_cur, pm, Ls[l].tmo, true);
    run_trimul(c, z_cur, pm, Ls[l].tmi, false);
    run_triatt(c, z_cur, pm, Ls[l].tas, true);
    run_triatt(c, z_cur, pm, Ls[l].tae, false);
    run_transition(c, z_cur, Ls[l].tz, c.xf16, RR);
    run_apb(c, s_cur, z_cur, mask, Ls[l].apb);
    run_transition(c, s_cur, Ls[l].ts, c.xs16, NT);
  }
}
